// MultiHeadAttention_57939108823264
// MI455X (gfx1250) — compile-verified, matrix operand order checked by reading, not run on hardware
//
#include <hip/hip_runtime.h>

// ---------------------------------------------------------------------------
// MHA forward for MI455X (gfx1250, wave32, WMMA 16x16x32 f16 -> f32 accum)
// ---------------------------------------------------------------------------
// BS=4, L=2048, D=1024, H=16, DK=64.  All matrix math through
// v_wmma_f32_16x16x32_f16.  Fragments follow CDNA5 ISA 7.12.2:
// 16-bit A/B frag for lane (mn = lane&15, half = lane>>4) is two contiguous
// 8-halfword runs at k0 + half*8 and k0 + 16 + half*8  -> two b128 loads.
// C/D layout: v8f element r <-> row (r + 8*half), column mn.
// Attention stages K/V tiles into LDS with double-buffered
// global_load_async_to_lds_b128 (ASYNCcnt) shared by all 4 waves of a block;
// softmax row-sums are computed on the matrix unit (P @ ones).

typedef __attribute__((ext_vector_type(16))) _Float16 v16h;
typedef __attribute__((ext_vector_type(8)))  _Float16 v8h;
typedef __attribute__((ext_vector_type(8)))  float    v8f;
typedef __attribute__((ext_vector_type(4)))  float    v4f;

typedef __attribute__((address_space(3))) char lds_char;

#define BS_  4
#define L_   2048
#define D_   1024
#define H_   16
#define DK_  64

__device__ __forceinline__ v8f wmma_f16(v16h a, v16h b, v8f c) {
  return __builtin_amdgcn_wmma_f32_16x16x32_f16(false, a, false, b,
                                                (short)0, c, false, false);
}

__device__ __forceinline__ v16h combine8(v8h lo, v8h hi) {
  v16h a;
#pragma unroll
  for (int i = 0; i < 8; ++i) { a[i] = lo[i]; a[i + 8] = hi[i]; }
  return a;
}

// f16 row-major fragment: row points at start of the row, k0 multiple of 32.
__device__ __forceinline__ v16h load_frag_f16(const _Float16* row, int k0, int half) {
  v8h lo = *(const v8h*)(row + k0 + half * 8);
  v8h hi = *(const v8h*)(row + k0 + 16 + half * 8);
  return combine8(lo, hi);
}

// fp32 row-major source, converted in-register to f16 fragment.
__device__ __forceinline__ v16h load_frag_f32(const float* row, int k0, int half) {
  const v4f* p = (const v4f*)(row + k0 + half * 8);
  v4f a0 = p[0], a1 = p[1];
  const v4f* q = (const v4f*)(row + k0 + 16 + half * 8);
  v4f b0 = q[0], b1 = q[1];
  v16h r;
#pragma unroll
  for (int i = 0; i < 4; ++i) {
    r[i]      = (_Float16)a0[i];
    r[4 + i]  = (_Float16)a1[i];
    r[8 + i]  = (_Float16)b0[i];
    r[12 + i] = (_Float16)b1[i];
  }
  return r;
}

// Async global -> LDS copy, 16 bytes per lane (ASYNCcnt-tracked).
__device__ __forceinline__ void async_ld_b128(unsigned lds_off, const void* g) {
  asm volatile("global_load_async_to_lds_b128 %0, %1, off"
               :: "v"(lds_off), "v"((unsigned long long)(uintptr_t)g)
               : "memory");
}

// ---------------------------------------------------------------------------
// Weight prep: Wt[n][k] = (f16) W[k][n]   (LDS-tiled transpose, 16x16)
// ---------------------------------------------------------------------------
__global__ __launch_bounds__(256)
void transpose_w_f16(const float* __restrict__ W, _Float16* __restrict__ Wt, int Dsz) {
  __shared__ float tile[16][17];
  int tx = threadIdx.x & 15;
  int ty = threadIdx.x >> 4;
  int bk = blockIdx.y * 16;  // k block
  int bn = blockIdx.x * 16;  // n block
  tile[ty][tx] = W[(size_t)(bk + ty) * Dsz + (bn + tx)];
  __syncthreads();
  Wt[(size_t)(bn + ty) * Dsz + (bk + tx)] = (_Float16)tile[tx][ty];
}

// ---------------------------------------------------------------------------
// Generic WMMA GEMM:  C[M,N] = A[M,K] @ Bt^T + bias
//   A: fp32 (AF32) or f16, row-major [M,K]
//   Bt: f16 [N,K] row-major (pre-transposed weights)
//   MODE 0: fp32 out[m*N+n]
//   MODE 1: f16 head-split     out[((b*H+h)*L + l)*DK + dk]
//   MODE 2: f16 head-split^T   out[((b*H+h)*DK + dk)*L + l]   (for V)
// Block: 256 thr = 8 waves (4 M-waves x 2 N-waves) -> 128M x 128N tile.
// Wave: 32M x 64N (8 accumulators; B fragment reused across both M subtiles,
// A fragments reused across all 4 N subtiles).
// ---------------------------------------------------------------------------
template <int MODE, bool AF32>
__global__ __launch_bounds__(256)
void gemm_wmma(const void* __restrict__ Aptr, const _Float16* __restrict__ Bt,
               const float* __restrict__ bias, void* __restrict__ outp,
               int M, int N, int K) {
  int lane  = threadIdx.x & 31;
  int wave  = threadIdx.x >> 5;
  int mn    = lane & 15;
  int half  = lane >> 4;
  int waveM = wave & 3;
  int waveN = wave >> 2;
  int m0 = blockIdx.y * 128 + waveM * 32;
  int n0 = blockIdx.x * 128 + waveN * 64;

  v8f acc[2][4] = {};

  const float*    A32r0 = (const float*)Aptr + (size_t)(m0 + mn) * K;
  const float*    A32r1 = (const float*)Aptr + (size_t)(m0 + 16 + mn) * K;
  const _Float16* A16r0 = (const _Float16*)Aptr + (size_t)(m0 + mn) * K;
  const _Float16* A16r1 = (const _Float16*)Aptr + (size_t)(m0 + 16 + mn) * K;
  const _Float16* Brow[4];
#pragma unroll
  for (int t = 0; t < 4; ++t)
    Brow[t] = Bt + (size_t)(n0 + 16 * t + mn) * K;

  for (int kk = 0; kk < K; kk += 32) {
    v16h a0, a1;
    if constexpr (AF32) {
      __builtin_prefetch(A32r0 + kk + 128, 0, 1);  // global_prefetch_b8
      __builtin_prefetch(A32r1 + kk + 128, 0, 1);
      a0 = load_frag_f32(A32r0, kk, half);
      a1 = load_frag_f32(A32r1, kk, half);
    } else {
      __builtin_prefetch(A16r0 + kk + 256, 0, 1);
      __builtin_prefetch(A16r1 + kk + 256, 0, 1);
      a0 = load_frag_f16(A16r0, kk, half);
      a1 = load_frag_f16(A16r1, kk, half);
    }
#pragma unroll
    for (int t = 0; t < 4; ++t) {
      v16h b = load_frag_f16(Brow[t], kk, half);
      acc[0][t] = wmma_f16(a0, b, acc[0][t]);
      acc[1][t] = wmma_f16(a1, b, acc[1][t]);
    }
  }

#pragma unroll
  for (int s = 0; s < 2; ++s) {
#pragma unroll
    for (int t = 0; t < 4; ++t) {
      int n = n0 + 16 * t + mn;
      float bv = bias[n];
#pragma unroll
      for (int r = 0; r < 8; ++r) {
        int m = m0 + 16 * s + r + 8 * half;
        float val = acc[s][t][r] + bv;
        if (MODE == 0) {
          ((float*)outp)[(size_t)m * N + n] = val;
        } else {
          int b  = m / L_;
          int l  = m - b * L_;
          int h  = n / DK_;
          int dk = n - h * DK_;
          _Float16* o = (_Float16*)outp;
          if (MODE == 1)
            o[((size_t)(b * H_ + h) * L_ + l) * DK_ + dk] = (_Float16)val;
          else
            o[((size_t)(b * H_ + h) * DK_ + dk) * L_ + l] = (_Float16)val;
        }
      }
    }
  }
}

// ---------------------------------------------------------------------------
// Flash attention: one wave per 16-query tile, keys in steps of 64.
// Q/K layout: [b,h,l,dk] f16 ; V layout: [b,h,dk,l] f16 (transposed).
// Block = 4 waves (128 thr), all on the same (b,h) -> K/V tiles are staged
// once per block into LDS via double-buffered global_load_async_to_lds_b128
// and shared by all 4 waves (4x less global traffic, ASYNCcnt overlap).
// Per 64-key iteration: 8 S-WMMAs, 2 rowsum-WMMAs, 8 PV-WMMAs.
// ---------------------------------------------------------------------------
__global__ __launch_bounds__(128)
void attn_wmma(const _Float16* __restrict__ Qh, const _Float16* __restrict__ Kh,
               const _Float16* __restrict__ Vt, const int* __restrict__ mask,
               _Float16* __restrict__ AO) {
  // kv[buf][0]=K tile [key][dk], kv[buf][1]=V tile [dk][key]  (32 KB)
  __shared__ __align__(16) _Float16 kvtile[2][2][64][64];
  __shared__ __align__(16) _Float16 plds[4][16][64];  // per-wave P tile (8 KB)

  int tid  = threadIdx.x;
  int lane = tid & 31;
  int wave = tid >> 5;
  int mn   = lane & 15;
  int half = lane >> 4;

  int bh = blockIdx.x;
  int b  = bh / H_;
  int h  = bh - b * H_;
  int q0 = blockIdx.y * 64 + wave * 16;

  const _Float16* Qp = Qh + (size_t)bh * L_ * DK_;
  const _Float16* Kp = Kh + (size_t)bh * L_ * DK_;
  const _Float16* Vp = Vt + (size_t)bh * DK_ * L_;
  const int*      mp = mask + (size_t)b * L_;

  // LDS byte offset of the staging buffer (ptrtoint of an LDS pointer).
  unsigned kvbase = (unsigned)(uintptr_t)(lds_char*)&kvtile[0][0][0][0];
  // byte offset helper: tile (buf, which) row-major [64][64] f16
  auto kvoff = [&](int buf, int which, int row, int col) -> unsigned {
    return kvbase + (unsigned)((((buf * 2 + which) * 64 + row) * 64 + col) * 2);
  };

  // Q fragments for this wave's 16 rows, pre-scaled by 1/sqrt(DK)=1/8 (exact).
  const _Float16* qrow = Qp + (size_t)(q0 + mn) * DK_;
  v16h aq0 = load_frag_f16(qrow, 0, half);
  v16h aq1 = load_frag_f16(qrow, 32, half);
  _Float16 sc = (_Float16)0.125f;
#pragma unroll
  for (int i = 0; i < 16; ++i) { aq0[i] *= sc; aq1[i] *= sc; }

  v16h ones;
#pragma unroll
  for (int i = 0; i < 16; ++i) ones[i] = (_Float16)1.0f;

  float rm[8];
#pragma unroll
  for (int r = 0; r < 8; ++r) rm[r] = -1e30f;
  v8f o[4] = {};
  v8f rsacc = {};  // per-row softmax denominator, via P @ ones

  // Prologue: stage first K/V tile (k0 = 0) into buffer 0.
  // 128 threads x 4 iters x 16B cover one 8 KB tile; K and V together.
#pragma unroll
  for (int it = 0; it < 4; ++it) {
    int i = tid + 128 * it;
    int row = i >> 3, seg = (i & 7) * 8;
    async_ld_b128(kvoff(0, 0, row, seg), Kp + (size_t)row * DK_ + seg);
    async_ld_b128(kvoff(0, 1, row, seg), Vp + (size_t)row * L_ + seg);
  }

  for (int k0 = 0; k0 < L_; k0 += 64) {
    int cur = (k0 >> 6) & 1;

    // ---- stage next tile into the other buffer, then sync current ----
    if (k0 + 64 < L_) {
      int nb = cur ^ 1, k0n = k0 + 64;
#pragma unroll
      for (int it = 0; it < 4; ++it) {
        int i = tid + 128 * it;
        int row = i >> 3, seg = (i & 7) * 8;
        async_ld_b128(kvoff(nb, 0, row, seg), Kp + (size_t)(k0n + row) * DK_ + seg);
        async_ld_b128(kvoff(nb, 1, row, seg), Vp + (size_t)row * L_ + k0n + seg);
      }
      // in-order completion: <=8 outstanding == only the just-issued prefetch
      asm volatile("s_wait_asynccnt 0x8" ::: "memory");
    } else {
      asm volatile("s_wait_asynccnt 0x0" ::: "memory");
    }
    __syncthreads();  // current buffer visible to all 4 waves

    const _Float16* Kt = &kvtile[cur][0][0][0];  // [key][dk]
    const _Float16* Vl = &kvtile[cur][1][0][0];  // [dk][key]

    // ---- S = Q @ K^T for 4 key subtiles (K fragments from LDS) ----
    v8f s[4] = {};
    int mk[4];
#pragma unroll
    for (int j = 0; j < 4; ++j) {
      const _Float16* kr = Kt + (size_t)(16 * j + mn) * 64;
      s[j] = wmma_f16(aq0, load_frag_f16(kr, 0, half), s[j]);
      s[j] = wmma_f16(aq1, load_frag_f16(kr, 32, half), s[j]);
      mk[j] = mp[k0 + 16 * j + mn];
    }
    // key-padding mask on columns (lane mn holds one key column per subtile)
#pragma unroll
    for (int j = 0; j < 4; ++j)
#pragma unroll
      for (int r = 0; r < 8; ++r) s[j][r] = mk[j] ? -1e30f : s[j][r];

    // ---- online softmax: row max via lane shuffles, then exp ----
    float corr[8];
#pragma unroll
    for (int r = 0; r < 8; ++r) {
      float mx = fmaxf(fmaxf(s[0][r], s[1][r]), fmaxf(s[2][r], s[3][r]));
#pragma unroll
      for (int off = 1; off < 16; off <<= 1)
        mx = fmaxf(mx, __shfl_xor(mx, off, 32));
      float nm = fmaxf(rm[r], mx);
      corr[r] = __expf(rm[r] - nm);
      rm[r] = nm;
#pragma unroll
      for (int j = 0; j < 4; ++j) s[j][r] = __expf(s[j][r] - nm);
    }
    // rescale running accumulators (O and the row-sum accumulator)
#pragma unroll
    for (int r = 0; r < 8; ++r) {
      rsacc[r] *= corr[r];
#pragma unroll
      for (int t = 0; t < 4; ++t) o[t][r] *= corr[r];
    }

    // ---- P: C-layout -> A-fragments via per-wave LDS tile (16 x 64 f16) ----
#pragma unroll
    for (int r = 0; r < 8; ++r) {
      int row = r + 8 * half;
#pragma unroll
      for (int j = 0; j < 4; ++j)
        plds[wave][row][16 * j + mn] = (_Float16)s[j][r];
    }
    asm volatile("s_wait_dscnt 0x0" ::: "memory");
    v16h pf0 = combine8(*(const v8h*)&plds[wave][mn][half * 8],
                        *(const v8h*)&plds[wave][mn][16 + half * 8]);
    v16h pf1 = combine8(*(const v8h*)&plds[wave][mn][32 + half * 8],
                        *(const v8h*)&plds[wave][mn][48 + half * 8]);
    asm volatile("" ::: "memory");

    // ---- row-sum of P on the matrix unit: rsacc += P @ ones ----
    rsacc = wmma_f16(pf0, ones, rsacc);
    rsacc = wmma_f16(pf1, ones, rsacc);

    // ---- O += P @ V  (V fragments from LDS, [dk][key] layout) ----
#pragma unroll
    for (int t = 0; t < 4; ++t) {
      const _Float16* vr = Vl + (size_t)(16 * t + mn) * 64;
      v16h bv0 = combine8(*(const v8h*)(vr + half * 8),
                          *(const v8h*)(vr + 16 + half * 8));
      v16h bv1 = combine8(*(const v8h*)(vr + 32 + half * 8),
                          *(const v8h*)(vr + 48 + half * 8));
      o[t] = wmma_f16(pf0, bv0, o[t]);
      o[t] = wmma_f16(pf1, bv1, o[t]);
    }

    __syncthreads();  // all waves done with current buffer before refill
  }

  // epilogue: O /= rowsum, store f16 [b, l, h*DK + dk]
#pragma unroll
  for (int t = 0; t < 4; ++t) {
#pragma unroll
    for (int r = 0; r < 8; ++r) {
      int l  = q0 + r + 8 * half;
      int dk = 16 * t + mn;
      float val = o[t][r] / rsacc[r];
      AO[((size_t)(b * L_ + l)) * D_ + h * DK_ + dk] = (_Float16)val;
    }
  }
}

// ---------------------------------------------------------------------------
// Host-side launch
// ---------------------------------------------------------------------------
extern "C" void kernel_launch(void* const* d_in, const int* in_sizes, int n_in,
                              void* d_out, int out_size, void* d_ws, size_t ws_size,
                              hipStream_t stream) {
  const float* x    = (const float*)d_in[0];
  const int*   mask = (const int*)  d_in[1];
  const float* Wq   = (const float*)d_in[2];
  const float* bq   = (const float*)d_in[3];
  const float* Wk   = (const float*)d_in[4];
  const float* bk   = (const float*)d_in[5];
  const float* Wv   = (const float*)d_in[6];
  const float* bv   = (const float*)d_in[7];
  const float* Wf   = (const float*)d_in[8];
  const float* bf   = (const float*)d_in[9];
  float* out = (float*)d_out;

  const int M = BS_ * L_;             // 8192
  const int N = D_;                   // 1024
  const int K = D_;                   // 1024
  const size_t DD = (size_t)D_ * D_;  // 1M elems
  const size_t MD = (size_t)M * D_;   // 8M elems

  // Workspace layout (f16): 4*DD + 4*MD = 36M halfwords = 72 MB
  _Float16* w   = (_Float16*)d_ws;
  _Float16* Wqt = w;
  _Float16* Wkt = Wqt + DD;
  _Float16* Wvt = Wkt + DD;
  _Float16* Wft = Wvt + DD;
  _Float16* Qh  = Wft + DD;
  _Float16* Kh  = Qh + MD;
  _Float16* Vth = Kh + MD;
  _Float16* AO  = Vth + MD;

  dim3 tgrid(D_ / 16, D_ / 16);
  transpose_w_f16<<<tgrid, 256, 0, stream>>>(Wq, Wqt, D_);
  transpose_w_f16<<<tgrid, 256, 0, stream>>>(Wk, Wkt, D_);
  transpose_w_f16<<<tgrid, 256, 0, stream>>>(Wv, Wvt, D_);
  transpose_w_f16<<<tgrid, 256, 0, stream>>>(Wf, Wft, D_);

  dim3 ggrid(N / 128, M / 128);  // (8, 64)
  gemm_wmma<1, true><<<ggrid, 256, 0, stream>>>(x, Wqt, bq, Qh,  M, N, K);
  gemm_wmma<1, true><<<ggrid, 256, 0, stream>>>(x, Wkt, bk, Kh,  M, N, K);
  gemm_wmma<2, true><<<ggrid, 256, 0, stream>>>(x, Wvt, bv, Vth, M, N, K);

  dim3 agrid(BS_ * H_, L_ / 64);  // (64, 32), 128 threads = 4 waves
  attn_wmma<<<agrid, 128, 0, stream>>>(Qh, Kh, Vth, mask, AO);

  gemm_wmma<0, false><<<ggrid, 256, 0, stream>>>(AO, Wft, bf, out, M, N, K);
}
